// NonLocalBlock2d_5995774345976
// MI455X (gfx1250) — matrix operand order repaired and checked by reading, not compiled, not run on hardware
//
#include <hip/hip_runtime.h>
#include <cstdint>
#include <cstddef>

// ---- problem dims (fixed by reference) ----
#define B_   8
#define C_   256
#define CI_  128
#define N_   4096     // H*W = 64*64
#define NP_  1024     // (H/2)*(W/2) = 32*32

typedef __attribute__((ext_vector_type(16))) __bf16 v16bf;
typedef __attribute__((ext_vector_type(8)))  __bf16 v8bf;
typedef __attribute__((ext_vector_type(8)))  float  v8f;

// Load 16 consecutive elements from global (fp32 or bf16) into a private
// bf16[16] (fp32 converts on the fly).
template <typename T>
__device__ __forceinline__ void load16regs(const T* __restrict__ src, __bf16* tmp) {
    if constexpr (sizeof(T) == 4) {
        const float4* s = reinterpret_cast<const float4*>(src);
#pragma unroll
        for (int i = 0; i < 4; ++i) {
            float4 v = s[i];
            tmp[4 * i + 0] = (__bf16)v.x;
            tmp[4 * i + 1] = (__bf16)v.y;
            tmp[4 * i + 2] = (__bf16)v.z;
            tmp[4 * i + 3] = (__bf16)v.w;
        }
    } else {
        const uint4* s = reinterpret_cast<const uint4*>(src);
        uint4 v0 = s[0];
        uint4 v1 = s[1];
        *reinterpret_cast<uint4*>(tmp)     = v0;
        *reinterpret_cast<uint4*>(tmp + 8) = v1;
    }
}

// 16 bf16 (32B) private -> LDS, as two b128 stores
__device__ __forceinline__ void store16lds(__bf16* dst, const __bf16* src) {
    *reinterpret_cast<uint4*>(dst)     = *reinterpret_cast<const uint4*>(src);
    *reinterpret_cast<uint4*>(dst + 8) = *reinterpret_cast<const uint4*>(src + 8);
}

// ---------------------------------------------------------------------------
// Generic bf16 WMMA GEMM:  C[M,N] = A[M,K] * B[K,N] (+ bias[m])
//   TA/TB in {float,__bf16} (converted to bf16 at LDS staging), TC in {float,__bf16}
//   BT   = true : B in memory is (K rows, N cols) -> transpose while staging
//          false: B in memory is (N rows, K cols) -> direct copy (p matrix case)
//   OUTT = true : store transposed, C[n*ldc + m]  (theta case)
// Block tile 128(M) x 256(N), BK=32, 256 threads = 8 waves (2x4),
// wave tile 64x64 = 16 WMMA accumulators. Double-buffered LDS, software
// pipelined: one barrier per K-step, global loads overlap the WMMA burst.
// ---------------------------------------------------------------------------
template <typename TA, typename TB, typename TC, bool BT, bool OUTT, bool BIAS>
__global__ void __launch_bounds__(256)
gemm_wmma(const TA* __restrict__ A, int lda, long long sA,
          const TB* __restrict__ Bm, int ldb, long long sB,
          TC* __restrict__ Cm, int ldc, long long sC,
          const float* __restrict__ bias, int K)
{
    __shared__ __bf16 Alds[2][128 * 32];   // [m][k]
    __shared__ __bf16 Blds[2][256 * 32];   // [n][k]  (transposed B tile)

    A  += (long long)blockIdx.z * sA;
    Bm += (long long)blockIdx.z * sB;
    Cm += (long long)blockIdx.z * sC;

    const int tid  = threadIdx.x;
    const int lane = tid & 31;
    const int wid  = tid >> 5;
    const int wm   = wid & 1;       // wave grid 2 (M) x 4 (N)
    const int wn   = wid >> 1;
    const int h    = lane >> 4;     // lane half
    const int ln   = lane & 15;

    const int blockM = blockIdx.y * 128;
    const int blockN = blockIdx.x * 256;

    // staging thread mapping
    const int arow = tid >> 1;             // A: 2 threads per m-row, 16 elems
    const int acb  = (tid & 1) * 16;
    const int krow = tid >> 3;             // B(BT): 8 threads per k-row, 32 n each
    const int nb   = (tid & 7) * 32;

    __bf16 ra[16] __attribute__((aligned(16)));
    __bf16 rb[32] __attribute__((aligned(16)));

    auto stage_load = [&](int kb) {
        load16regs<TA>(A + (long long)(blockM + arow) * lda + kb + acb, ra);
        if constexpr (BT) {
            const TB* bsrc = Bm + (long long)(kb + krow) * ldb + blockN + nb;
            load16regs<TB>(bsrc, rb);
            load16regs<TB>(bsrc + 16, rb + 16);
        } else {
            const TB* bsrc = Bm + (long long)(blockN + tid) * ldb + kb;
            load16regs<TB>(bsrc, rb);
            load16regs<TB>(bsrc + 16, rb + 16);
        }
    };
    auto stage_store = [&](int buf) {
        store16lds(&Alds[buf][arow * 32 + acb], ra);
        if constexpr (BT) {
#pragma unroll
            for (int e = 0; e < 32; ++e)
                Blds[buf][(nb + e) * 32 + krow] = rb[e];
        } else {
            store16lds(&Blds[buf][tid * 32], rb);
            store16lds(&Blds[buf][tid * 32 + 16], rb + 16);
        }
    };

    v8f acc[4][4] = {};

    auto compute = [&](int buf) {
        // A fragments (contiguous K layout: half picks K{0-15}/{16-31})
        v16bf af[4];
#pragma unroll
        for (int i = 0; i < 4; ++i) {
            const __bf16* ap = &Alds[buf][(wm * 64 + i * 16 + ln) * 32];
            v8bf a0 = *reinterpret_cast<const v8bf*>(ap + h * 16);
            v8bf a1 = *reinterpret_cast<const v8bf*>(ap + h * 16 + 8);
#pragma unroll
            for (int e = 0; e < 8; ++e) { af[i][e] = a0[e]; af[i][e + 8] = a1[e]; }
        }
#pragma unroll
        for (int j = 0; j < 4; ++j) {
            v16bf bf = *reinterpret_cast<const v16bf*>(
                &Blds[buf][(wn * 64 + j * 16 + ln) * 32 + h * 16]);
#pragma unroll
            for (int i = 0; i < 4; ++i) {
                acc[i][j] = __builtin_amdgcn_wmma_f32_16x16x32_bf16(
                    false, af[i], false, bf, (short)0, acc[i][j], false, false);
            }
        }
    };

    // ---- software pipeline: prologue ----
    int cur = 0;
    stage_load(0);
    stage_store(0);
    __syncthreads();

    for (int kb = 32; kb < K; kb += 32) {
        // prefetch the K+64 tiles (speculative; OOB prefetch is dropped)
        __builtin_prefetch(A + (long long)(blockM + arow) * lda + kb + 32 + acb, 0, 1);
        if constexpr (BT)
            __builtin_prefetch(Bm + (long long)(kb + 32 + krow) * ldb + blockN + nb, 0, 1);
        else
            __builtin_prefetch(Bm + (long long)(blockN + tid) * ldb + kb + 32, 0, 1);

        stage_load(kb);          // global loads for next tile (overlap compute)
        compute(cur);            // WMMA burst from current buffer
        stage_store(cur ^ 1);    // publish next tile
        __syncthreads();
        cur ^= 1;
    }
    compute(cur);                // epilogue

    // ---- store D (VGPR r -> M = r + 8*half, lanes -> N) ----
#pragma unroll
    for (int i = 0; i < 4; ++i) {
#pragma unroll
        for (int j = 0; j < 4; ++j) {
            int col = blockN + wn * 64 + j * 16 + ln;
#pragma unroll
            for (int r = 0; r < 8; ++r) {
                int row = blockM + wm * 64 + i * 16 + r + 8 * h;
                float v = acc[i][j][r];
                if constexpr (BIAS) v += bias[row];
                if constexpr (OUTT) Cm[(long long)col * ldc + row] = (TC)v;
                else                Cm[(long long)row * ldc + col] = (TC)v;
            }
        }
    }
}

// ---------------------------------------------------------------------------
// 2x2 max pool, stride 2: (B,C,64,64) -> (B,C,32,32)
// ---------------------------------------------------------------------------
__global__ void pool2x2(const float* __restrict__ x, float* __restrict__ xp, int total)
{
    int idx = blockIdx.x * blockDim.x + threadIdx.x;
    if (idx >= total) return;
    int w2 = idx & 31;
    int h2 = (idx >> 5) & 31;
    int bc = idx >> 10;   // b*C + c
    const float* p = x + ((long long)bc * 64 + h2 * 2) * 64 + w2 * 2;
    float m0 = fmaxf(p[0], p[1]);
    float m1 = fmaxf(p[64], p[65]);
    xp[idx] = fmaxf(m0, m1);
}

// ---------------------------------------------------------------------------
// Row softmax over 1024 elements, in place, with 1/sqrt(CI) scale folded in.
// ---------------------------------------------------------------------------
__global__ void __launch_bounds__(256) softmax_rows(float* __restrict__ s)
{
    __shared__ float red[256];
    float* p = s + (long long)blockIdx.x * NP_;
    int tid = threadIdx.x;
    float4 v = reinterpret_cast<float4*>(p)[tid];

    float m = fmaxf(fmaxf(v.x, v.y), fmaxf(v.z, v.w));
    red[tid] = m; __syncthreads();
#pragma unroll
    for (int st = 128; st > 0; st >>= 1) {
        if (tid < st) red[tid] = fmaxf(red[tid], red[tid + st]);
        __syncthreads();
    }
    float rowmax = red[0];
    __syncthreads();

    const float scale = 0.08838834764831845f;   // 1/sqrt(128)
    float4 e;
    e.x = __expf((v.x - rowmax) * scale);
    e.y = __expf((v.y - rowmax) * scale);
    e.z = __expf((v.z - rowmax) * scale);
    e.w = __expf((v.w - rowmax) * scale);

    red[tid] = e.x + e.y + e.z + e.w; __syncthreads();
#pragma unroll
    for (int st = 128; st > 0; st >>= 1) {
        if (tid < st) red[tid] += red[tid + st];
        __syncthreads();
    }
    float inv = 1.0f / red[0];
    e.x *= inv; e.y *= inv; e.z *= inv; e.w *= inv;
    reinterpret_cast<float4*>(p)[tid] = e;
}

// ---------------------------------------------------------------------------
// BatchNorm training-mode stats: one block per channel over B*N elements.
// stats[c] = mean, stats[C_+c] = rsqrt(var + eps)
// ---------------------------------------------------------------------------
__global__ void __launch_bounds__(256) bn_stats(const float* __restrict__ z,
                                                float* __restrict__ stats)
{
    __shared__ float rs[256], rq[256];
    int c = blockIdx.x;
    int tid = threadIdx.x;
    float s = 0.f, q = 0.f;
    for (int i = tid; i < B_ * N_; i += 256) {
        int b = i >> 12;
        int n = i & (N_ - 1);
        float v = z[((long long)b * C_ + c) * N_ + n];
        s += v;
        q += v * v;
    }
    rs[tid] = s; rq[tid] = q; __syncthreads();
#pragma unroll
    for (int st = 128; st > 0; st >>= 1) {
        if (tid < st) { rs[tid] += rs[tid + st]; rq[tid] += rq[tid + st]; }
        __syncthreads();
    }
    if (tid == 0) {
        const float inv_n = 1.0f / (float)(B_ * N_);
        float mean = rs[0] * inv_n;
        float var  = rq[0] * inv_n - mean * mean;
        stats[c]      = mean;
        stats[C_ + c] = rsqrtf(var + 1e-5f);
    }
}

// ---------------------------------------------------------------------------
// out = x + gamma * (z - mean) * invstd + beta      (residual fused)
// ---------------------------------------------------------------------------
__global__ void bn_apply(const float* __restrict__ x, const float* __restrict__ z,
                         const float* __restrict__ stats,
                         const float* __restrict__ gamma, const float* __restrict__ beta,
                         float* __restrict__ out, long long total)
{
    long long idx = (long long)blockIdx.x * blockDim.x + threadIdx.x;
    if (idx >= total) return;
    int c = (int)((idx >> 12) & (C_ - 1));
    float zn = (z[idx] - stats[c]) * stats[C_ + c];
    out[idx] = x[idx] + zn * gamma[c] + beta[c];
}

// ---------------------------------------------------------------------------
extern "C" void kernel_launch(void* const* d_in, const int* in_sizes, int n_in,
                              void* d_out, int out_size, void* d_ws, size_t ws_size,
                              hipStream_t stream)
{
    const float* x       = (const float*)d_in[0];
    const float* theta_w = (const float*)d_in[1];
    const float* theta_b = (const float*)d_in[2];
    const float* phi_w   = (const float*)d_in[3];
    const float* phi_b   = (const float*)d_in[4];
    const float* g_w     = (const float*)d_in[5];
    const float* g_b     = (const float*)d_in[6];
    const float* wz_w    = (const float*)d_in[7];
    const float* wz_b    = (const float*)d_in[8];
    const float* bn_g    = (const float*)d_in[9];
    const float* bn_b    = (const float*)d_in[10];
    float* out = (float*)d_out;

    // ---- workspace carve (~188 MB) ----
    char* wsp = (char*)d_ws;
    auto carve = [&](size_t bytes) -> char* {
        char* p = wsp;
        wsp += (bytes + 255) & ~(size_t)255;
        return p;
    };
    float*  xp      = (float*) carve((size_t)B_ * C_  * NP_ * 4);  // pooled x
    __bf16* theta_t = (__bf16*)carve((size_t)B_ * N_  * CI_ * 2);  // theta^T (N,CI)
    __bf16* phiB    = (__bf16*)carve((size_t)B_ * CI_ * NP_ * 2);  // phi (CI,N')
    __bf16* gB      = (__bf16*)carve((size_t)B_ * CI_ * NP_ * 2);  // g   (CI,N')
    float*  scores  = (float*) carve((size_t)B_ * N_  * NP_ * 4);  // scores / p in place
    __bf16* tB      = (__bf16*)carve((size_t)B_ * CI_ * N_  * 2);  // t (CI,N)
    float*  z       = (float*) carve((size_t)B_ * C_  * N_  * 4);  // z (C,N)
    float*  stats   = (float*) carve((size_t)2 * C_ * 4);

    // 1) 2x2 max pool
    {
        int total = B_ * C_ * NP_;
        pool2x2<<<(total + 255) / 256, 256, 0, stream>>>(x, xp, total);
    }

    // 2) theta = theta_w @ x[b]  (128x4096, K=256), out transposed bf16 (4096x128)
    gemm_wmma<float, float, __bf16, true, true, true>
        <<<dim3(N_ / 256, CI_ / 128, B_), 256, 0, stream>>>(
            theta_w, C_, 0,
            x, N_, (long long)C_ * N_,
            theta_t, CI_, (long long)N_ * CI_,
            theta_b, C_);

    // 3) phi = phi_w @ xp[b]  (128x1024, K=256)
    gemm_wmma<float, float, __bf16, true, false, true>
        <<<dim3(NP_ / 256, CI_ / 128, B_), 256, 0, stream>>>(
            phi_w, C_, 0,
            xp, NP_, (long long)C_ * NP_,
            phiB, NP_, (long long)CI_ * NP_,
            phi_b, C_);

    // 4) g = g_w @ xp[b]
    gemm_wmma<float, float, __bf16, true, false, true>
        <<<dim3(NP_ / 256, CI_ / 128, B_), 256, 0, stream>>>(
            g_w, C_, 0,
            xp, NP_, (long long)C_ * NP_,
            gB, NP_, (long long)CI_ * NP_,
            g_b, C_);

    // 5) scores = theta^T @ phi  (4096x1024, K=128), fp32 out
    gemm_wmma<__bf16, __bf16, float, true, false, false>
        <<<dim3(NP_ / 256, N_ / 128, B_), 256, 0, stream>>>(
            theta_t, CI_, (long long)N_ * CI_,
            phiB, NP_, (long long)CI_ * NP_,
            scores, NP_, (long long)N_ * NP_,
            nullptr, CI_);

    // 6) softmax over last dim (scale folded), in place
    softmax_rows<<<B_ * N_, 256, 0, stream>>>(scores);

    // 7) t = g @ p^T  (128x4096, K=1024); p is (N,N') -> BN mode (direct stage)
    gemm_wmma<__bf16, float, __bf16, false, false, false>
        <<<dim3(N_ / 256, CI_ / 128, B_), 256, 0, stream>>>(
            gB, NP_, (long long)CI_ * NP_,
            scores, NP_, (long long)N_ * NP_,
            tB, N_, (long long)CI_ * N_,
            nullptr, NP_);

    // 8) z = wz_w @ t + wz_b  (256x4096, K=128), fp32 out
    gemm_wmma<float, __bf16, float, true, false, true>
        <<<dim3(N_ / 256, C_ / 128, B_), 256, 0, stream>>>(
            wz_w, CI_, 0,
            tB, N_, (long long)CI_ * N_,
            z, N_, (long long)C_ * N_,
            wz_b, CI_);

    // 9) batch-norm statistics (one block per channel)
    bn_stats<<<C_, 256, 0, stream>>>(z, stats);

    // 10) normalize + affine + residual
    {
        long long total = (long long)B_ * C_ * N_;
        bn_apply<<<(int)((total + 255) / 256), 256, 0, stream>>>(
            x, z, stats, bn_g, bn_b, out, total);
    }
}
